// RpnTarget_17789754541026
// MI455X (gfx1250) — compile-verified, matrix operand order checked by reading, not run on hardware
//
#include <hip/hip_runtime.h>
#include <hip/hip_bf16.h>

// ---------------- CDNA5 feature detection ----------------
#if defined(__HIP_DEVICE_COMPILE__) && defined(__gfx1250__) && \
    __has_builtin(__builtin_amdgcn_global_load_async_to_lds_b128)
#define HAVE_ASYNC 1
#else
#define HAVE_ASYNC 0
#endif

#if HAVE_ASYNC
#if __has_builtin(__builtin_amdgcn_s_wait_asynccnt)
#define WAIT_ASYNC(N) __builtin_amdgcn_s_wait_asynccnt(N)
#else
#define WAIT_ASYNC(N) asm volatile("s_wait_asynccnt %0" ::"i"(N) : "memory")
#endif
#else
#define WAIT_ASYNC(N) do {} while (0)
#endif

typedef __attribute__((ext_vector_type(16))) _Float16 v16h;
typedef __attribute__((ext_vector_type(8)))  float    v8f;

// The async-LDS builtins take pointers to a 16-byte int vector:
//   arg0: AS1 (global) v4i*,  arg1: AS3 (LDS) v4i*,  imm offset, imm cpol
typedef int v4i_ __attribute__((vector_size(16)));
typedef __attribute__((address_space(1))) v4i_* as1v4p;
typedef __attribute__((address_space(3))) v4i_* as3v4p;

// ---------------- problem constants ----------------
#define GBOX          8
#define MAXB          4
#define TRAIN_ANCHORS 800
#define POS_THR       0.5f
#define NEG_THR       0.02f

#define CHUNK         512         // anchors per LDS chunk
#define SCAN_BLOCKS   512
#define NEG_CAP       4096        // power of two (bitonic)
#define POS_CAP       1024

// workspace layout (bytes)
#define WS_NEGCNT   0                               // 16 x u32
#define WS_POSCNT   64                              // 16 x u32
#define WS_NEGTOT   128                             // 16 x f32
#define WS_NEGCAND  256                             // MAXB*NEG_CAP x u64
#define WS_POSCAND  (256 + MAXB * NEG_CAP * 8)      // MAXB*POS_CAP x uint4

// ---------------- helpers ----------------
__device__ __forceinline__ float rnd01(unsigned x) {
  x ^= x >> 17; x *= 0xed5ad4bbu;
  x ^= x >> 11; x *= 0xac4c1b51u;
  x ^= x >> 15; x *= 0x31848babu;
  x ^= x >> 14;
  return (float)(x >> 8) * (1.0f / 16777216.0f);
}

// 16B global -> LDS, async (CDNA5 GLOBAL_LOAD_ASYNC_TO_LDS_B128) with fallback
__device__ __forceinline__ void async_cp16(const float* g, float* l) {
#if HAVE_ASYNC
  const v4i_* pg = (const v4i_*)g;   // pointee-type cast (generic)
  v4i_*       pl = (v4i_*)l;
  __builtin_amdgcn_global_load_async_to_lds_b128(
      (as1v4p)pg, (as3v4p)pl, 0, 0); // explicit addrspace casts
#else
  float4 t = *(const float4*)g;
  *(float4*)l = t;
#endif
}

// ---------------- kernel 1: zero outputs + ws header ----------------
__global__ void rpn_init(int* __restrict__ out, long long n, unsigned* __restrict__ hdr) {
  long long stride = (long long)gridDim.x * blockDim.x;
  long long i0 = (long long)blockIdx.x * blockDim.x + threadIdx.x;
  int4 z; z.x = 0; z.y = 0; z.z = 0; z.w = 0;
  long long n4 = n >> 2;
  for (long long v = i0; v < n4; v += stride) ((int4*)out)[v] = z;
  if (i0 < (n & 3)) out[n4 * 4 + i0] = 0;
  if (i0 < 64) hdr[i0] = 0;
}

// ---------------- kernel 2: streaming IoU scan ----------------
__global__ __launch_bounds__(256) void rpn_scan(
    const float* __restrict__ anchors, const float* __restrict__ gtb,
    int A, int Bimg, unsigned char* __restrict__ ws) {
  __shared__ float sA[2][CHUNK * 6];      // double-buffered anchor chunks (24 KB)
  __shared__ float sBox[MAXB * GBOX * 6]; // all gt boxes

  unsigned*           negCnt  = (unsigned*)(ws + WS_NEGCNT);
  unsigned*           posCnt  = (unsigned*)(ws + WS_POSCNT);
  float*              negTot  = (float*)(ws + WS_NEGTOT);
  unsigned long long* negCand = (unsigned long long*)(ws + WS_NEGCAND);
  uint4*              posCand = (uint4*)(ws + WS_POSCAND);

  const int tid = threadIdx.x;
  const int Bc  = (Bimg < MAXB) ? Bimg : MAXB;
  const size_t nFloats = (size_t)A * 6;
  const int nChunks = (A + CHUNK - 1) / CHUNK;
  const float negPriThr = 1.0f - fminf(0.5f, 2048.0f / (float)A);

  // prologue: kick off async fill of buffer 0
  if (blockIdx.x < nChunks) {
    size_t base = (size_t)blockIdx.x * (CHUNK * 6);
#pragma unroll
    for (int t = 0; t < 3; ++t) {
      size_t fo = base + (size_t)(tid + t * 256) * 4;
      if (fo + 4 > nFloats) fo = nFloats - 4;  // clamp tail (stays in-bounds)
      async_cp16(anchors + fo, &sA[0][(tid + t * 256) * 4]);
    }
  }
  for (int i = tid; i < Bc * GBOX * 6; i += 256) sBox[i] = gtb[i];

  float negLocal[MAXB] = {0.f, 0.f, 0.f, 0.f};

  int buf = 0;
  for (int c = blockIdx.x; c < nChunks; c += gridDim.x) {
    int cn = c + gridDim.x;
    if (cn < nChunks) {  // prefetch next chunk into the other buffer
      size_t base = (size_t)cn * (CHUNK * 6);
#pragma unroll
      for (int t = 0; t < 3; ++t) {
        size_t fo = base + (size_t)(tid + t * 256) * 4;
        if (fo + 4 > nFloats) fo = nFloats - 4;
        async_cp16(anchors + fo, &sA[buf ^ 1][(tid + t * 256) * 4]);
      }
      WAIT_ASYNC(3);     // current chunk's 3 transfers done; next 3 in flight
    } else {
      WAIT_ASYNC(0);
    }
    __syncthreads();

    int na = A - c * CHUNK; if (na > CHUNK) na = CHUNK;
    for (int l = tid; l < na; l += 256) {
      const int aidx = c * CHUNK + l;
      const float* a6 = &sA[buf][l * 6];
      float a0 = a6[0], a1 = a6[1], a2 = a6[2], a3 = a6[3], a4 = a6[4], a5 = a6[5];
      float va = (a3 - a0) * (a4 - a1) * (a5 - a2);
      for (int b = 0; b < Bc; ++b) {
        float best = -1.0f; int bg = 0;
#pragma unroll
        for (int g = 0; g < GBOX; ++g) {
          const float* q = &sBox[(b * GBOX + g) * 6];
          float i0 = fminf(a3, q[3]) - fmaxf(a0, q[0]);
          float i1 = fminf(a4, q[4]) - fmaxf(a1, q[1]);
          float i2 = fminf(a5, q[5]) - fmaxf(a2, q[2]);
          float inter = fmaxf(i0, 0.f) * fmaxf(i1, 0.f) * fmaxf(i2, 0.f);
          float vb = (q[3] - q[0]) * (q[4] - q[1]) * (q[5] - q[2]);
          float iou = inter / (vb + va - inter);
          if (iou > best) { best = iou; bg = g; }
        }
        float pri = rnd01((unsigned)(b * A) + (unsigned)aidx);
        if (best >= POS_THR) {
          unsigned slot = atomicAdd(&posCnt[b], 1u);
          if (slot < POS_CAP) {
            uint4 r; r.x = __float_as_uint(pri); r.y = (unsigned)aidx;
            r.z = (unsigned)bg; r.w = 0u;
            posCand[(size_t)b * POS_CAP + slot] = r;
          }
        } else if (best <= NEG_THR) {
          negLocal[b] += 1.0f;
          if (pri > negPriThr) {  // superset of global top-800 w.h.p.
            unsigned slot = atomicAdd(&negCnt[b], 1u);
            if (slot < NEG_CAP)
              negCand[(size_t)b * NEG_CAP + slot] =
                  ((unsigned long long)__float_as_uint(pri) << 32) | (unsigned)aidx;
          }
        }
      }
    }
    __syncthreads();  // buffer may be re-filled next iteration
    buf ^= 1;
  }

  // ---- wave-level reduction of per-lane negative counts via the matrix unit
  // A[lane] has count in element 0; B = ones  =>  D[m][n] = cnt[m] + cnt[m+16].
  // Lanes 0..15 hold rows 0..7, lanes 16..31 rows 8..15; lane 0 and lane 16
  // each sum their 8 rows and the two atomic adds combine to the wave total.
  v16h ones;
#pragma unroll
  for (int i = 0; i < 16; ++i) ones[i] = (_Float16)1.0f;
  for (int b = 0; b < Bc; ++b) {
    v16h av;
#pragma unroll
    for (int i = 0; i < 16; ++i) av[i] = (_Float16)0.0f;
    av[0] = (_Float16)negLocal[b];
    v8f cz = {};
    v8f d = __builtin_amdgcn_wmma_f32_16x16x32_f16(
        false, av, false, ones, (short)0, cz, false, false);
    float s = d[0] + d[1] + d[2] + d[3] + d[4] + d[5] + d[6] + d[7];
    if ((threadIdx.x & 15u) == 0u) atomicAdd(&negTot[b], s);
  }
}

// ---------------- kernel 3: per-image top-k selection + writeback ----------
__global__ __launch_bounds__(1024) void rpn_select(
    const float* __restrict__ anchors, const float* __restrict__ gtb,
    int A, int Bimg, unsigned char* __restrict__ ws,
    int* __restrict__ tags, float* __restrict__ deltas) {
  __shared__ unsigned long long kneg[NEG_CAP];  // 32 KB
  __shared__ unsigned long long rk[1024];
  __shared__ int ri[1024];
  __shared__ int s_posnum;

  const int b = blockIdx.x;
  const int tid = threadIdx.x;
  (void)Bimg;

  unsigned*           negCnt  = (unsigned*)(ws + WS_NEGCNT);
  unsigned*           posCnt  = (unsigned*)(ws + WS_POSCNT);
  float*              negTot  = (float*)(ws + WS_NEGTOT);
  unsigned long long* negCand = (unsigned long long*)(ws + WS_NEGCAND);
  uint4*              posCand = (uint4*)(ws + WS_POSCAND);

  int ncnt = (int)negCnt[b]; if (ncnt > NEG_CAP) ncnt = NEG_CAP;
  int pcnt = (int)posCnt[b]; if (pcnt > POS_CAP) pcnt = POS_CAP;

  const unsigned long long* srcNeg = negCand + (size_t)b * NEG_CAP;
  for (int i = tid; i < NEG_CAP; i += 1024)
    kneg[i] = (i < ncnt) ? srcNeg[i] : 0ull;

  // ---- positives: top-2 via two max reductions, winner thread writes output
  unsigned long long mykey = 0ull; unsigned mygt = 0u;
  if (tid < pcnt) {
    uint4 r = posCand[(size_t)b * POS_CAP + tid];
    mykey = ((unsigned long long)r.x << 32) | (unsigned long long)r.y;
    mygt = r.z;
  }
  if (tid == 0) s_posnum = 0;
  for (int pass = 0; pass < 2; ++pass) {
    __syncthreads();
    rk[tid] = mykey; ri[tid] = tid;
    for (int s = 512; s > 0; s >>= 1) {
      __syncthreads();
      if (tid < s && rk[tid + s] > rk[tid]) { rk[tid] = rk[tid + s]; ri[tid] = ri[tid + s]; }
    }
    __syncthreads();
    if (rk[0] != 0ull) {
      if (tid == 0) s_posnum++;
      if (tid == ri[0]) {
        unsigned aidx = (unsigned)(mykey & 0xffffffffu);
        const float* a6 = anchors + (size_t)aidx * 6;
        const float* g6 = gtb + ((size_t)b * GBOX + mygt) * 6;
        float as0 = a6[3] - a6[0], as1 = a6[4] - a6[1], as2 = a6[5] - a6[2];
        float ac0 = (a6[0] + a6[3]) * 0.5f, ac1 = (a6[1] + a6[4]) * 0.5f,
              ac2 = (a6[2] + a6[5]) * 0.5f;
        float gs0 = g6[3] - g6[0], gs1 = g6[4] - g6[1], gs2 = g6[5] - g6[2];
        float gc0 = (g6[0] + g6[3]) * 0.5f, gc1 = (g6[1] + g6[4]) * 0.5f,
              gc2 = (g6[2] + g6[5]) * 0.5f;
        float* dp = deltas + ((size_t)b * A + aidx) * 6;
        dp[0] = (gc0 - ac0) / as0; dp[1] = (gc1 - ac1) / as1; dp[2] = (gc2 - ac2) / as2;
        dp[3] = logf(gs0 / as0);   dp[4] = logf(gs1 / as1);   dp[5] = logf(gs2 / as2);
        tags[(size_t)b * A + aidx] = 1;
        mykey = 0ull;  // exclude from pass 2
      }
    }
  }
  __syncthreads();
  int posnum = s_posnum;

  // ---- negatives: in-LDS bitonic sort (descending by priority), take quota
  for (int k = 2; k <= NEG_CAP; k <<= 1)
    for (int j = k >> 1; j > 0; j >>= 1) {
      __syncthreads();
      for (int i = tid; i < NEG_CAP; i += 1024) {
        int ixj = i ^ j;
        if (ixj > i) {
          unsigned long long x = kneg[i], y = kneg[ixj];
          bool desc = ((i & k) == 0);
          if ((x < y) == desc) { kneg[i] = y; kneg[ixj] = x; }
        }
      }
    }
  __syncthreads();

  int quota = TRAIN_ANCHORS - posnum; if (quota < 0) quota = 0;
  int take = (ncnt < quota) ? ncnt : quota;
  int tot = (int)negTot[b]; if (take > tot) take = tot;  // WMMA-reduced count
  for (int i = tid; i < take; i += 1024) {
    unsigned aidx = (unsigned)(kneg[i] & 0xffffffffu);
    tags[(size_t)b * A + aidx] = -1;
  }
}

// ---------------- host launcher ----------------
extern "C" void kernel_launch(void* const* d_in, const int* in_sizes, int n_in,
                              void* d_out, int out_size, void* d_ws, size_t ws_size,
                              hipStream_t stream) {
  (void)ws_size;
  const float* anchors = (const float*)d_in[0];
  const float* gtb     = (const float*)d_in[1];
  // d_in[2] = gt_labels: unused by the reference forward.

  int A = in_sizes[0] / 6;
  int Bimg = (n_in > 2 ? in_sizes[2] : MAXB * GBOX) / GBOX;
  if (Bimg > MAXB) Bimg = MAXB;
  if (Bimg < 1) Bimg = 1;

  int*   tags   = (int*)d_out;
  float* deltas = (float*)d_out + (size_t)Bimg * A;
  unsigned char* ws = (unsigned char*)d_ws;

  rpn_init<<<1024, 256, 0, stream>>>((int*)d_out, (long long)out_size, (unsigned*)d_ws);
  rpn_scan<<<SCAN_BLOCKS, 256, 0, stream>>>(anchors, gtb, A, Bimg, ws);
  rpn_select<<<Bimg, 1024, 0, stream>>>(anchors, gtb, A, Bimg, ws, tags, deltas);
}
